// MambaMixer_60962765799477
// MI455X (gfx1250) — matrix operand order repaired and checked by reading, not compiled, not run on hardware
//
#include <hip/hip_runtime.h>
#include <hip/hip_bf16.h>

typedef __bf16 bf16;
typedef __attribute__((ext_vector_type(16))) __bf16 v16bf;
typedef __attribute__((ext_vector_type(8)))  __bf16 v8bf;
typedef __attribute__((ext_vector_type(8)))  float  v8f;
typedef __attribute__((ext_vector_type(4)))  float  v4f;

#define DEV static __device__ __forceinline__

DEV v8f wmma_bf16(v16bf a, v16bf b, v8f c) {
  return __builtin_amdgcn_wmma_f32_16x16x32_bf16(false, a, false, b, (short)0, c, false, false);
}

union frag_u { v16bf v; v8bf h[2]; };

// ---- WMMA fragment loaders (contiguous k = 16h + s order, wave32) ----
// All are contiguous per lane -> two b128 loads each (four for the fp32 source).

// A fragment, 16x32 tile from row-major source: lane holds A[m][16h..16h+15]
DEV v16bf ldA(const bf16* p, int ld) {
  int lane = threadIdx.x & 31;
  int m = lane & 15, hf = lane >> 4;
  const bf16* q = p + m * ld + 16 * hf;
  frag_u u;
  u.h[0] = *(const v8bf*)(q);
  u.h[1] = *(const v8bf*)(q + 8);
  return u.v;
}

// A fragment with per-row (m) scale rs[m].
DEV v16bf ldA_rs(const bf16* p, int ld, const float* rs) {
  int lane = threadIdx.x & 31;
  int m = lane & 15, hf = lane >> 4;
  float s = rs[m];
  const bf16* q = p + m * ld + 16 * hf;
  v8bf lo = *(const v8bf*)(q);
  v8bf hi = *(const v8bf*)(q + 8);
  v16bf a;
#pragma unroll
  for (int i = 0; i < 8; ++i) {
    a[i]     = (bf16)((float)lo[i] * s);
    a[8 + i] = (bf16)((float)hi[i] * s);
  }
  return a;
}

// B fragment where B[k][c] = p[c*ld + k]: lane holds p[c][16h..16h+15]
DEV v16bf ldBt(const bf16* p, int ld) {
  int lane = threadIdx.x & 31;
  int c = lane & 15, hf = lane >> 4;
  const bf16* q = p + c * ld + 16 * hf;
  frag_u u;
  u.h[0] = *(const v8bf*)(q);
  u.h[1] = *(const v8bf*)(q + 8);
  return u.v;
}

// Same from contiguous fp32 source (converted on the fly).
DEV v16bf ldBt_f32(const float* p, int ld) {
  int lane = threadIdx.x & 31;
  int c = lane & 15, hf = lane >> 4;
  const float* q = p + c * ld + 16 * hf;
  v4f f0 = *(const v4f*)(q);
  v4f f1 = *(const v4f*)(q + 4);
  v4f f2 = *(const v4f*)(q + 8);
  v4f f3 = *(const v4f*)(q + 12);
  v16bf b;
#pragma unroll
  for (int i = 0; i < 4; ++i) {
    b[i]      = (bf16)f0[i];
    b[4 + i]  = (bf16)f1[i];
    b[8 + i]  = (bf16)f2[i];
    b[12 + i] = (bf16)f3[i];
  }
  return b;
}

// ------------------- conversion kernels -------------------

__global__ void cvt_f32_bf16(const float* __restrict__ in, bf16* __restrict__ out, long long n) {
  long long i = (long long)blockIdx.x * blockDim.x + threadIdx.x;
  if (i < n) out[i] = (bf16)in[i];
}

// in_proj_w (5152 x 2048) -> bf16, rows padded with zeros to 5248
__global__ void cvt_w1_pad(const float* __restrict__ in, bf16* __restrict__ out) {
  long long i = (long long)blockIdx.x * blockDim.x + threadIdx.x;
  if (i >= 5248LL * 2048LL) return;
  int r = (int)(i / 2048);
  out[i] = (r < 5152) ? (bf16)in[i] : (bf16)0.0f;
}

// ------------------- NT GEMM: C[M x N] = A[M x K] * W[N x K]^T -------------------
// block = 256 threads = 8 waves in a 2(M) x 4(N) grid; wave tile 32x32; block tile 64x128.
__global__ void __launch_bounds__(256) gemm_nt(const bf16* __restrict__ A,
                                               const bf16* __restrict__ W,
                                               float* __restrict__ C,
                                               int K, int ldc) {
  int w = threadIdx.x >> 5;
  int wm = w >> 2, wn = w & 3;
  long long m0 = (long long)blockIdx.y * 64 + wm * 32;
  long long n0 = (long long)blockIdx.x * 128 + wn * 32;
  v8f acc[2][2] = {};
  const bf16* Ap = A + m0 * K;
  const bf16* Wp = W + n0 * K;
  for (int k = 0; k < K; k += 32) {
    v16bf a0 = ldA(Ap + k, K);
    v16bf a1 = ldA(Ap + 16LL * K + k, K);
    v16bf b0 = ldBt(Wp + k, K);
    v16bf b1 = ldBt(Wp + 16LL * K + k, K);
    acc[0][0] = wmma_bf16(a0, b0, acc[0][0]);
    acc[0][1] = wmma_bf16(a0, b1, acc[0][1]);
    acc[1][0] = wmma_bf16(a1, b0, acc[1][0]);
    acc[1][1] = wmma_bf16(a1, b1, acc[1][1]);
  }
  int lane = threadIdx.x & 31, nn = lane & 15, mb = (lane >> 4) * 8;
#pragma unroll
  for (int i = 0; i < 2; ++i)
#pragma unroll
    for (int j = 0; j < 2; ++j)
#pragma unroll
      for (int r = 0; r < 8; ++r)
        C[(m0 + i * 16 + mb + r) * ldc + (n0 + j * 16 + nn)] = acc[i][j][r];
}

// ------------------- causal depthwise conv + SiLU -------------------
// zx: 8192 x 5248 (xBC at col 2048..5120); xact: 8192 x 3072
__global__ void conv_silu(const float* __restrict__ zx, const float* __restrict__ cw,
                          const float* __restrict__ cb, float* __restrict__ xact) {
  long long i = (long long)blockIdx.x * blockDim.x + threadIdx.x;
  if (i >= 8192LL * 3072LL) return;
  int c = (int)(i % 3072);
  long long bt = i / 3072;
  int t = (int)(bt % 4096);
  long long brow = bt - t;  // b*4096
  float acc = cb[c];
#pragma unroll
  for (int k = 0; k < 4; ++k) {
    int tt = t - 3 + k;
    if (tt >= 0) acc += zx[(brow + tt) * 5248 + 2048 + c] * cw[c * 4 + k];
  }
  xact[i] = acc / (1.0f + expf(-acc));  // silu
}

// dt = softplus(zx[..., 5120 + h] + dt_bias[h]); dtb: 8192 x 32
__global__ void dt_softplus(const float* __restrict__ zx, const float* __restrict__ dt_bias,
                            float* __restrict__ dtb) {
  long long i = (long long)blockIdx.x * blockDim.x + threadIdx.x;
  if (i >= 8192LL * 32LL) return;
  int h = (int)(i & 31);
  long long bt = i >> 5;
  float x = zx[bt * 5248 + 5120 + h] + dt_bias[h];
  dtb[i] = (x > 20.0f) ? x : log1pf(expf(x));
}

// ------------------- SSD core: one block per (batch, head) -------------------
// Internal chunk = 64; 64 sequential chunks carry state (64x64 fp32) in LDS.
// All WMMA fragment loads are lane-contiguous b128 LDS reads.
__global__ void __launch_bounds__(256) ssd_kernel(const float* __restrict__ xact,
                                                  const float* __restrict__ dtb,
                                                  const float* __restrict__ A_log,
                                                  const float* __restrict__ Dvec,
                                                  float* __restrict__ ybuf) {
  int bh = blockIdx.x;
  int b = bh >> 5, h = bh & 31, g = h >> 2;
  __shared__ float s_dt[64], s_acum[64], s_eacum[64], s_decay[64];
  __shared__ __attribute__((aligned(16))) bf16 sC[64 * 64];    // C[l][n]
  __shared__ __attribute__((aligned(16))) bf16 sB[64 * 64];    // B[l][n]
  __shared__ __attribute__((aligned(16))) bf16 sXT[64 * 64];   // (x*dt)^T : [p][l]
  __shared__ __attribute__((aligned(16))) bf16 sBdT[64 * 64];  // (B*decay)^T : [n][l]
  __shared__ __attribute__((aligned(16))) bf16 sS[64 * 64];    // masked S[l][s]
  __shared__ __attribute__((aligned(16))) float s_state[64 * 64];  // state[p][n]

  int tid = threadIdx.x;
  int w = tid >> 5, lane = tid & 31, nn = lane & 15, mb = (lane >> 4) * 8;
  float Ah = -expf(A_log[h]);
  float Dh = Dvec[h];
  for (int i = tid; i < 4096; i += 256) s_state[i] = 0.0f;
  __syncthreads();

  for (int ic = 0; ic < 64; ++ic) {
    long long base = (long long)b * 4096 + ic * 64;
    if (tid < 64) s_dt[tid] = dtb[(base + tid) * 32 + h];
    __syncthreads();
    if (tid == 0) {
      float a = 0.0f;
      for (int l = 0; l < 64; ++l) { a += s_dt[l] * Ah; s_acum[l] = a; }
    }
    __syncthreads();
    if (tid < 64) {
      s_eacum[tid] = expf(s_acum[tid]);
      s_decay[tid] = expf(s_acum[63] - s_acum[tid]);
    }
    __syncthreads();
    // stage C, B, X^T*dt, (B*decay)^T as bf16 tiles
    for (int e = tid; e < 4096; e += 256) {
      int l = e >> 6, n = e & 63;
      long long row = (base + l) * 3072;
      float xv = xact[row + h * 64 + n];
      float bv = xact[row + 2048 + g * 64 + n];
      float cv = xact[row + 2560 + g * 64 + n];
      sXT[n * 64 + l]  = (bf16)(xv * s_dt[l]);
      sB[e]            = (bf16)bv;
      sBdT[n * 64 + l] = (bf16)(bv * s_decay[l]);
      sC[e]            = (bf16)cv;
    }
    __syncthreads();

    // S[l][s] = (C . B^T) * Lmat -> bf16 LDS
#pragma unroll
    for (int j = 0; j < 2; ++j) {
      int t2 = w * 2 + j, ti = t2 >> 2, tj = t2 & 3;
      v8f acc = {};
      for (int ks = 0; ks < 64; ks += 32) {
        v16bf a  = ldA(sC + ti * 16 * 64 + ks, 64);
        v16bf bb = ldBt(sB + tj * 16 * 64 + ks, 64);
        acc = wmma_bf16(a, bb, acc);
      }
      int scol = tj * 16 + nn;
#pragma unroll
      for (int r = 0; r < 8; ++r) {
        int l = ti * 16 + mb + r;
        float val = (l >= scol) ? expf(s_acum[l] - s_acum[scol]) * acc[r] : 0.0f;
        sS[l * 64 + scol] = (bf16)val;
      }
    }
    __syncthreads();

    // Y = S @ X  +  (C * e^{acum}) @ state^T ; then + D*x, store
#pragma unroll
    for (int j = 0; j < 2; ++j) {
      int t2 = w * 2 + j, ti = t2 >> 2, tj = t2 & 3;
      v8f acc = {};
      for (int ks = 0; ks < 64; ks += 32) {
        v16bf a  = ldA(sS + ti * 16 * 64 + ks, 64);            // S[l][s]
        v16bf bb = ldBt(sXT + tj * 16 * 64 + ks, 64);          // X[s][p] = sXT[p][s]
        acc = wmma_bf16(a, bb, acc);
      }
      for (int ks = 0; ks < 64; ks += 32) {
        v16bf a  = ldA_rs(sC + ti * 16 * 64 + ks, 64, s_eacum + ti * 16);
        v16bf bb = ldBt_f32(s_state + (tj * 16) * 64 + ks, 64);  // state[p][n]
        acc = wmma_bf16(a, bb, acc);
      }
      int p = tj * 16 + nn;
#pragma unroll
      for (int r = 0; r < 8; ++r) {
        int l = ti * 16 + mb + r;
        float x0 = xact[(base + l) * 3072 + h * 64 + p];
        ybuf[(base + l) * 2048 + h * 64 + p] = acc[r] + Dh * x0;
      }
    }

    // newstate[p][n] = sum_l X[l][p] * (B[l][n] * decay[l])
    v8f ns[2];
#pragma unroll
    for (int j = 0; j < 2; ++j) {
      int t2 = w * 2 + j, ti = t2 >> 2, tj = t2 & 3;
      v8f acc = {};
      for (int ks = 0; ks < 64; ks += 32) {
        v16bf a  = ldA(sXT + ti * 16 * 64 + ks, 64);           // A[p][l] = sXT row-major
        v16bf bb = ldBt(sBdT + tj * 16 * 64 + ks, 64);         // B[l][n] = sBdT[n][l]
        acc = wmma_bf16(a, bb, acc);
      }
      ns[j] = acc;
    }
    __syncthreads();  // all reads of s_state done before update
    float dchunk = s_eacum[63];
#pragma unroll
    for (int j = 0; j < 2; ++j) {
      int t2 = w * 2 + j, ti = t2 >> 2, tj = t2 & 3;
      int n = tj * 16 + nn;
#pragma unroll
      for (int r = 0; r < 8; ++r) {
        int p = ti * 16 + mb + r;
        s_state[p * 64 + n] = s_state[p * 64 + n] * dchunk + ns[j][r];
      }
    }
    __syncthreads();
  }
}

// ------------------- gating + grouped RMSNorm -> bf16 -------------------
// One block per token; wave w handles group w (256 channels, 8 per lane).
__global__ void __launch_bounds__(256) gate_norm(const float* __restrict__ ybuf,
                                                 const float* __restrict__ zx,
                                                 const float* __restrict__ norm_w,
                                                 bf16* __restrict__ ynorm) {
  long long t = blockIdx.x;
  int w = threadIdx.x >> 5, lane = threadIdx.x & 31;
  int cbase = w * 256 + lane * 8;
  float vals[8];
  float ss = 0.0f;
#pragma unroll
  for (int i = 0; i < 8; ++i) {
    int c = cbase + i;
    float y = ybuf[t * 2048 + c];
    float z = zx[t * 5248 + c];
    float v = y * (z / (1.0f + expf(-z)));
    vals[i] = v;
    ss += v * v;
  }
  for (int m = 16; m >= 1; m >>= 1) ss += __shfl_xor(ss, m, 32);
  float rs = rsqrtf(ss / 256.0f + 1e-5f);
#pragma unroll
  for (int i = 0; i < 8; ++i) {
    int c = cbase + i;
    ynorm[t * 2048 + c] = (bf16)(vals[i] * rs * norm_w[c]);
  }
}

// ------------------- host launch -------------------

extern "C" void kernel_launch(void* const* d_in, const int* in_sizes, int n_in,
                              void* d_out, int out_size, void* d_ws, size_t ws_size,
                              hipStream_t stream) {
  const float* hs      = (const float*)d_in[0];  // (2,4096,2048)
  const float* w_in    = (const float*)d_in[1];  // (5152,2048)
  const float* conv_w  = (const float*)d_in[2];  // (3072,4)
  const float* conv_b  = (const float*)d_in[3];  // (3072,)
  const float* dt_bias = (const float*)d_in[4];  // (32,)
  const float* A_log   = (const float*)d_in[5];  // (32,)
  const float* Dv      = (const float*)d_in[6];  // (32,)
  const float* norm_w  = (const float*)d_in[7];  // (2048,)
  const float* w_out   = (const float*)d_in[8];  // (2048,2048)
  float* out = (float*)d_out;                    // (2,4096,2048)

  char* ws = (char*)d_ws;
  bf16*  hs_bf = (bf16*)(ws + 0);                 // 8192x2048 bf16 (reused as ynorm)
  bf16*  w1_bf = (bf16*)(ws + 33554432ull);       // 5248x2048 bf16 (zero-padded)
  bf16*  w2_bf = (bf16*)(ws + 55050240ull);       // 2048x2048 bf16
  float* zx    = (float*)(ws + 63438848ull);      // 8192x5248 f32
  float* xact  = (float*)(ws + 235405312ull);     // 8192x3072 f32
  float* dtb   = (float*)(ws + 336068608ull);     // 8192x32 f32
  float* ybuf  = (float*)(ws + 337117184ull);     // 8192x2048 f32
  bf16*  ynorm = hs_bf;                            // alias (hs_bf dead after gemm1)

  cvt_f32_bf16<<<65536, 256, 0, stream>>>(hs, hs_bf, 16777216LL);
  cvt_w1_pad<<<41984, 256, 0, stream>>>(w_in, w1_bf);
  cvt_f32_bf16<<<16384, 256, 0, stream>>>(w_out, w2_bf, 4194304LL);

  // zxBCdt = hs @ in_proj_w^T  (M=8192, N=5248 padded, K=2048)
  gemm_nt<<<dim3(41, 128), 256, 0, stream>>>(hs_bf, w1_bf, zx, 2048, 5248);

  conv_silu<<<98304, 256, 0, stream>>>(zx, conv_w, conv_b, xact);
  dt_softplus<<<1024, 256, 0, stream>>>(zx, dt_bias, dtb);

  ssd_kernel<<<64, 256, 0, stream>>>(xact, dtb, A_log, Dv, ybuf);

  gate_norm<<<8192, 256, 0, stream>>>(ybuf, zx, norm_w, ynorm);

  // out = ynorm @ out_proj_w^T  (M=8192, N=2048, K=2048)
  gemm_nt<<<dim3(16, 128), 256, 0, stream>>>(ynorm, w2_bf, out, 2048, 2048);
}
